// AngleProtoLoss_11527692223243
// MI455X (gfx1250) — matrix operand order repaired and checked by reading, not compiled, not run on hardware
//
#include <hip/hip_runtime.h>
#include <hip/hip_bf16.h>
#include <math.h>

#define N_SPK 8192
#define M_UTT 10
#define D_EMB 128
#define CLAMP_MIN 1e-6f
#define TILES (N_SPK / 16)   // 512 column/row tiles
#define NW 8                 // waves per block (wave32 -> 256 threads)
#define ITERS (TILES / NW)   // 64 uniform iterations per wave

typedef __attribute__((ext_vector_type(16))) __bf16 v16bf;
typedef __attribute__((ext_vector_type(8)))  __bf16 v8bf;
typedef __attribute__((ext_vector_type(8)))  float  v8f;

// ---------------------------------------------------------------------------
// Kernel 1: one wave per speaker. centroid = mean over M_UTT, last = x[:,9,:],
// L2-normalize both, store bf16 row-major [N_SPK][D_EMB] for WMMA consumption.
// ---------------------------------------------------------------------------
__global__ void __launch_bounds__(256)
prep_kernel(const float* __restrict__ x,
            __hip_bfloat16* __restrict__ Abf,   // normalized last rows
            __hip_bfloat16* __restrict__ Bbf)   // normalized centroids
{
    int spk  = (blockIdx.x * blockDim.x + threadIdx.x) >> 5;
    int lane = threadIdx.x & 31;
    if (spk >= N_SPK) return;

    const float* xr = x + (size_t)spk * (M_UTT * D_EMB);

    float cent[4], last[4];
    float ssc = 0.f, ssl = 0.f;
#pragma unroll
    for (int q = 0; q < 4; ++q) {
        int d = lane + 32 * q;                 // coalesced across the wave
        float acc = 0.f;
#pragma unroll
        for (int m = 0; m < M_UTT; ++m) acc += xr[m * D_EMB + d];
        cent[q] = acc * (1.0f / (float)M_UTT);
        last[q] = xr[(M_UTT - 1) * D_EMB + d];
        ssc += cent[q] * cent[q];
        ssl += last[q] * last[q];
    }
#pragma unroll
    for (int off = 16; off >= 1; off >>= 1) {
        ssc += __shfl_xor(ssc, off, 32);
        ssl += __shfl_xor(ssl, off, 32);
    }
    float rc = rsqrtf(ssc);
    float rl = rsqrtf(ssl);
#pragma unroll
    for (int q = 0; q < 4; ++q) {
        int d = lane + 32 * q;
        Bbf[(size_t)spk * D_EMB + d] = __float2bfloat16(cent[q] * rc);
        Abf[(size_t)spk * D_EMB + d] = __float2bfloat16(last[q] * rl);
    }
}

// B-tile fragment set for one 16-column tile (4 x K=32 slices).
struct BTile { v16bf f[4]; };

__device__ __forceinline__ BTile load_btile(const __bf16* __restrict__ Bm,
                                            int j, int N, int bbase)
{
    BTile t;
    const __bf16* brow = Bm + (size_t)(j * 16 + N) * D_EMB + bbase;
#pragma unroll
    for (int kb = 0; kb < 4; ++kb) {
        v8bf lo = *(const v8bf*)(brow + kb * 32);
        v8bf hi = *(const v8bf*)(brow + kb * 32 + 16);
        t.f[kb] = __builtin_shufflevector(lo, hi,
                   0,1,2,3,4,5,6,7,8,9,10,11,12,13,14,15);
    }
    return t;
}

// ---------------------------------------------------------------------------
// Kernel 2: fused bf16-WMMA GEMM + streaming log-softmax + diagonal NLL.
// Block = one 16-row tile (blockIdx.x = r); 8 waves stride the 512 col tiles.
// Software-pipelined: B fragments for tile j+NW are prefetched into a second
// register buffer while tile j's WMMAs + exp tail execute, so s_wait_loadcnt
// covers a full iteration of slack instead of stalling before every WMMA.
// logits <= K = b + |w| (cos clipped into [1e-6, ~1]) -> fixed-shift sum-exp.
// ---------------------------------------------------------------------------
__global__ void __launch_bounds__(NW * 32)
gemm_lse_kernel(const __hip_bfloat16* __restrict__ Abf,
                const __hip_bfloat16* __restrict__ Bbf,
                const float* __restrict__ wp,
                const float* __restrict__ bp,
                float* __restrict__ out)
{
    __shared__ float lds_s[NW][16];
    __shared__ float lds_diag[16];

    const int tid  = threadIdx.x;
    const int lane = tid & 31;
    const int wid  = tid >> 5;
    const int r    = blockIdx.x;               // row tile index

    const float W  = wp[0];
    const float Bc = bp[0];
    const float K  = Bc + fabsf(W);            // upper bound on any logit

    const __bf16* A  = (const __bf16*)Abf;
    const __bf16* Bm = (const __bf16*)Bbf;

    // ---- A fragments for row tile r (ISA 16-bit A 16x32 layout), in regs.
    v16bf afrag[4];
    {
        const int M     = lane & 15;
        const int abase = (lane >= 16) ? 8 : 0;
        const __bf16* arow = A + (size_t)(r * 16 + M) * D_EMB + abase;
#pragma unroll
        for (int kb = 0; kb < 4; ++kb) {
            v8bf lo = *(const v8bf*)(arow + kb * 32);
            v8bf hi = *(const v8bf*)(arow + kb * 32 + 16);
            afrag[kb] = __builtin_shufflevector(lo, hi,
                         0,1,2,3,4,5,6,7,8,9,10,11,12,13,14,15);
        }
    }

    float s[8];
#pragma unroll
    for (int i = 0; i < 8; ++i) s[i] = 0.f;

    // B layout (32x16 KxN, 16-bit): lane = N (&15); lanes<16 hold K 0..7,16..23,
    // lanes>=16 hold K 8..15,24..31 (per kb), matching the A layout.
    const int  N        = lane & 15;
    const int  bbase    = (lane >= 16) ? 8 : 0;
    const bool diagLane = (lane < 8) || (lane >= 24);
    const int  dsel     = lane & 7;
    const int  drow     = dsel + ((lane >= 16) ? 8 : 0);

    BTile cur = load_btile(Bm, wid, N, bbase);  // prologue prefetch

    for (int it = 0; it < ITERS; ++it) {
        const int j  = wid + it * NW;
        // Branchless wrap on the last iteration: one redundant L2-hit load
        // keeps the loop body straight-line (preserves clause pairing).
        int jn = j + NW;
        if (jn >= TILES) jn = wid;
        BTile nxt = load_btile(Bm, jn, N, bbase);

        v8f c = {};
#pragma unroll
        for (int kb = 0; kb < 4; ++kb) {
            c = __builtin_amdgcn_wmma_f32_16x16x32_bf16(
                    /*neg_a=*/false, afrag[kb],
                    /*neg_b=*/false, cur.f[kb],
                    /*c_mod=*/(short)0, c,
                    /*reuse_a=*/false, /*reuse_b=*/false);
        }

        const bool diagTile = (j == r);
#pragma unroll
        for (int i = 0; i < 8; ++i) {
            float lg = fmaxf(c[i], CLAMP_MIN) * W + Bc;
            // D-layout: diag (m,m) sits in lanes {0..7,24..31} at vgpr lane&7
            if (diagTile && diagLane && (i == dsel)) lds_diag[drow] = lg;
            s[i] += __expf(lg - K);
        }

        cur = nxt;
    }

    // reduce sum-exp across the 16 lanes of each half (rows 0-7 / 8-15)
#pragma unroll
    for (int off = 1; off < 16; off <<= 1) {
#pragma unroll
        for (int i = 0; i < 8; ++i) s[i] += __shfl_xor(s[i], off, 32);
    }
    if ((lane & 15) == 0) {
        const int rbase = (lane >> 4) * 8;
#pragma unroll
        for (int i = 0; i < 8; ++i) lds_s[wid][rbase + i] = s[i];
    }
    __syncthreads();

    if (tid < 16) {
        float S = 0.f;
#pragma unroll
        for (int w2 = 0; w2 < NW; ++w2) S += lds_s[w2][tid];
        const float lse     = K + __logf(S);
        const float contrib = (lse - lds_diag[tid]) * (1.0f / (float)N_SPK);
        atomicAdd(out, contrib);   // out zeroed via hipMemsetAsync each call
    }
}

// ---------------------------------------------------------------------------
extern "C" void kernel_launch(void* const* d_in, const int* in_sizes, int n_in,
                              void* d_out, int out_size, void* d_ws, size_t ws_size,
                              hipStream_t stream)
{
    const float* x = (const float*)d_in[0];
    const float* w = (const float*)d_in[1];
    const float* b = (const float*)d_in[2];
    float* out = (float*)d_out;

    __hip_bfloat16* Abf = (__hip_bfloat16*)d_ws;                     // 2 MB
    __hip_bfloat16* Bbf = Abf + (size_t)N_SPK * D_EMB;               // 2 MB

    hipMemsetAsync(out, 0, sizeof(float), stream);
    prep_kernel<<<N_SPK / 8, 256, 0, stream>>>(x, Abf, Bbf);
    gemm_lse_kernel<<<TILES, NW * 32, 0, stream>>>(Abf, Bbf, w, b, out);
}
